// MFG4ADGenerator_17497696764530
// MI455X (gfx1250) — matrix operand order repaired and checked by reading, not compiled, not run on hardware
//
#include <hip/hip_runtime.h>
#include <hip/hip_bf16.h>

typedef __attribute__((ext_vector_type(16))) _Float16 v16h;
typedef __attribute__((ext_vector_type(8)))  float    v8f;

#define FH 64
#define MHID 512
#define DT_C 0.1f
#define BETA_DT_C 0.05f
#define K_TAYLOR_C 40

// ---------------- utility kernels ----------------

__global__ void zero_kernel(float* __restrict__ p, int n) {
    int i = blockIdx.x * blockDim.x + threadIdx.x;
    if (i < n) p[i] = 0.0f;
}

__global__ void init_term_acc(const float* __restrict__ tau,
                              float* __restrict__ term, float* __restrict__ acc, int n) {
    int i = blockIdx.x * blockDim.x + threadIdx.x;
    if (i < n) { float t = tau[i]; term[i] = t; acc[i] = t; }
}

// deg = segment_sum(edge_weight, row)
__global__ void deg_kernel(const int* __restrict__ row, const float* __restrict__ ew,
                           float* __restrict__ deg, int ne) {
    int e = blockIdx.x * blockDim.x + threadIdx.x;
    if (e < ne) atomicAdd(&deg[row[e]], ew[e]);
}

// s[row] += ew * term[col]
__global__ void spmv_edge(const int* __restrict__ row, const int* __restrict__ col,
                          const float* __restrict__ ew, const float* __restrict__ term,
                          float* __restrict__ s, int ne) {
    int e = blockIdx.x * blockDim.x + threadIdx.x;
    if (e < ne) atomicAdd(&s[row[e]], ew[e] * term[col[e]]);
}

// term = coef*(deg*term - s);  acc += term
__global__ void taylor_node(const float* __restrict__ deg, const float* __restrict__ s,
                            float* __restrict__ term, float* __restrict__ acc,
                            float coef, int n) {
    int i = blockIdx.x * blockDim.x + threadIdx.x;
    if (i >= n) return;
    float nt = coef * (deg[i] * term[i] - s[i]);
    term[i] = nt;
    acc[i] += nt;
}

// ---------------- feature MLP ----------------

// layer1 raw: out[n,j] = gamma*tau[n]*fw1[0,j] + lam*P[n]*fw1[1,j] + fb1[j]  (Q==0)
__global__ void layer1_kernel(const float* __restrict__ tau, const float* __restrict__ P,
                              const float* __restrict__ gamma, const float* __restrict__ lam,
                              const float* __restrict__ fw1, const float* __restrict__ fb1,
                              float* __restrict__ out, int n) {
    int i = blockIdx.x * blockDim.x + threadIdx.x;
    int node = i >> 6, j = i & 63;
    if (node >= n) return;
    float f0 = gamma[0] * tau[node];
    float f1 = lam[0] * P[node];
    out[i] = f0 * fw1[j] + f1 * fw1[FH + j] + fb1[j];
}

// layernorm(64) + relu, one wave (32 lanes) per row, 8 rows per 256-thread block
__global__ void ln_relu_64(const float* __restrict__ in, const float* __restrict__ g,
                           const float* __restrict__ bta, float* __restrict__ out, int n) {
    int wave = threadIdx.x >> 5;
    int lane = threadIdx.x & 31;
    int row = blockIdx.x * 8 + wave;
    if (row >= n) return;
    float a = in[row * FH + lane];
    float b = in[row * FH + lane + 32];
    float s = a + b;
    #pragma unroll
    for (int off = 16; off; off >>= 1) s += __shfl_xor(s, off, 32);
    float mean = s * (1.0f / 64.0f);
    float da = a - mean, db = b - mean;
    float v = da * da + db * db;
    #pragma unroll
    for (int off = 16; off; off >>= 1) v += __shfl_xor(v, off, 32);
    float rstd = rsqrtf(v * (1.0f / 64.0f) + 1e-5f);
    float oa = da * rstd * g[lane] + bta[lane];
    float ob = db * rstd * g[lane + 32] + bta[lane + 32];
    out[row * FH + lane]      = fmaxf(oa, 0.0f);
    out[row * FH + lane + 32] = fmaxf(ob, 0.0f);
}

// hidden GEMM: (N x 64) @ (64 x 64) + fb2  via v_wmma_f32_16x16x32_f16
// block = 128 threads (4 waves); each wave computes a 16x64 output tile; block covers 64 rows.
// Activations and weights are staged in LDS as f16 (convert once during the cooperative load).
__global__ void gemm_fw2_wmma(const float* __restrict__ hin, const float* __restrict__ fw2,
                              const float* __restrict__ fb2, float* __restrict__ hout, int n) {
    __shared__ _Float16 sA[64 * 64];     // activation tile, row-major, f16
    __shared__ _Float16 sW[64 * 64];     // weights, K-major rows: sW[k*64 + j], f16
    int tid = threadIdx.x;
    int rowBase = blockIdx.x * 64;
    bool fullTile = (rowBase + 64 <= n);

    if (fullTile) {
        for (int idx = tid; idx < 64 * 64; idx += 128)
            sA[idx] = (_Float16)hin[rowBase * FH + idx];
    } else {
        for (int idx = tid; idx < 64 * 64; idx += 128) {
            int gr = rowBase + (idx >> 6);
            sA[idx] = (gr < n) ? (_Float16)hin[gr * FH + (idx & 63)] : (_Float16)0.0f;
        }
    }
    for (int idx = tid; idx < 64 * 64; idx += 128)
        sW[idx] = (_Float16)fw2[idx];
    __syncthreads();

    int wave = tid >> 5;
    int lane = tid & 31;
    int half = lane >> 4;   // lane-half selects K sub-blocks (A) / K groups (B), M offset (D)
    int mn   = lane & 15;
    int r0   = wave * 16;   // wave's 16-row slab inside the 64-row block tile

    // A fragments (16x32 f16, ISA 7.12.2 layout): element e -> vgpr v=e/2,
    // K = [v<4 ? 2v : 16+2(v-4)] + (e&1) + half*8, M = lane&15
    v16h afrag[2];
    #pragma unroll
    for (int kt = 0; kt < 2; ++kt) {
        #pragma unroll
        for (int e = 0; e < 16; ++e) {
            int v = e >> 1;
            int kbase = (v < 4) ? (v * 2) : (16 + (v - 4) * 2);
            int K = kt * 32 + kbase + (e & 1) + half * 8;
            afrag[kt][e] = sA[(r0 + mn) * 64 + K];
        }
    }

    #pragma unroll
    for (int nt = 0; nt < 4; ++nt) {
        int colj = nt * 16 + mn;
        float fb2v = fb2[colj];          // hoisted bias (invariant over r)
        v8f c = {};
        #pragma unroll
        for (int kt = 0; kt < 2; ++kt) {
            // B fragment (32x16): element e holds K = kt*32 + half*8 + e + (e>=8 ? 8 : 0), N = lane&15
            v16h b;
            #pragma unroll
            for (int e = 0; e < 16; ++e) {
                int K = kt * 32 + half * 8 + e + ((e >= 8) ? 8 : 0);
                b[e] = sW[K * 64 + colj];
            }
            c = __builtin_amdgcn_wmma_f32_16x16x32_f16(
                    false, afrag[kt], false, b, (short)0, c, false, false);
        }
        // D layout: vgpr r holds M = r + half*8, N = lane&15
        int rowW = rowBase + r0 + half * 8;   // rows rowW .. rowW+7 for r=0..7
        if (fullTile) {
            #pragma unroll
            for (int r = 0; r < 8; ++r)
                hout[(rowW + r) * FH + colj] = c[r] + fb2v;
        } else {
            #pragma unroll
            for (int r = 0; r < 8; ++r)
                if (rowW + r < n) hout[(rowW + r) * FH + colj] = c[r] + fb2v;
        }
    }
}

// nu = h @ fwo + fbo ; u_nu = tau * nu   (3 outputs per node)
__global__ void layer_out_kernel(const float* __restrict__ h, const float* __restrict__ fwo,
                                 const float* __restrict__ fbo, const float* __restrict__ tau,
                                 float* __restrict__ u_nu, int n) {
    int node = blockIdx.x * blockDim.x + threadIdx.x;
    if (node >= n) return;
    float a0 = fbo[0], a1 = fbo[1], a2 = fbo[2];
    #pragma unroll 4
    for (int k = 0; k < FH; ++k) {
        float hv = h[node * FH + k];
        a0 += hv * fwo[k * 3 + 0];
        a1 += hv * fwo[k * 3 + 1];
        a2 += hv * fwo[k * 3 + 2];
    }
    float t = tau[node];
    u_nu[node * 3 + 0] = t * a0;
    u_nu[node * 3 + 1] = t * a1;
    u_nu[node * 3 + 2] = t * a2;
}

// div[row] += (u_nu[row]-u_nu[col])*ew  (per component)
__global__ void flux_div_kernel(const int* __restrict__ row, const int* __restrict__ col,
                                const float* __restrict__ ew, const float* __restrict__ u_nu,
                                float* __restrict__ divv, int ne) {
    int e = blockIdx.x * blockDim.x + threadIdx.x;
    if (e >= ne) return;
    int r = row[e], c = col[e];
    float w = ew[e];
    atomicAdd(&divv[r * 3 + 0], (u_nu[r * 3 + 0] - u_nu[c * 3 + 0]) * w);
    atomicAdd(&divv[r * 3 + 1], (u_nu[r * 3 + 1] - u_nu[c * 3 + 1]) * w);
    atomicAdd(&divv[r * 3 + 2], (u_nu[r * 3 + 2] - u_nu[c * 3 + 2]) * w);
}

// hat_u = tau + DT*(-sqrt(|div|^2+1e-12) + P)   (Q==0)
__global__ void hat_kernel(const float* __restrict__ tau, const float* __restrict__ P,
                           const float* __restrict__ divv, float* __restrict__ hat, int n) {
    int i = blockIdx.x * blockDim.x + threadIdx.x;
    if (i >= n) return;
    float d0 = divv[i * 3 + 0], d1 = divv[i * 3 + 1], d2 = divv[i * 3 + 2];
    float ds = sqrtf(d0 * d0 + d1 * d1 + d2 * d2 + 1e-12f);
    hat[i] = tau[i] + DT_C * (-ds + P[i]);
}

// pre_hm[j] += sum over a 256-row slab of x[i]*mw1[i,j]; x = [hat_u, P] (Q block is zero -> skipped)
__global__ void gemv1_kernel(const float* __restrict__ hat, const float* __restrict__ P,
                             const float* __restrict__ mw1, float* __restrict__ pre, int n) {
    int j = threadIdx.x;            // 512 threads == 512 columns
    int i0 = blockIdx.x * 256;
    int iend = i0 + 256;
    int lim = 2 * n;
    if (iend > lim) iend = lim;
    float acc = 0.0f;
    for (int i = i0; i < iend; ++i) {
        float xv = (i < n) ? hat[i] : P[i - n];
        acc += xv * mw1[(size_t)i * MHID + j];
    }
    atomicAdd(&pre[j], acc);
}

// layernorm(512)+relu with bias mb1 added first; single 512-thread block
__global__ void ln_relu_512(const float* __restrict__ pre, const float* __restrict__ mb1,
                            const float* __restrict__ g, const float* __restrict__ bta,
                            float* __restrict__ out) {
    __shared__ float red[MHID];
    int t = threadIdx.x;
    float x = pre[t] + mb1[t];
    red[t] = x;
    __syncthreads();
    for (int off = 256; off; off >>= 1) { if (t < off) red[t] += red[t + off]; __syncthreads(); }
    float mean = red[0] * (1.0f / MHID);
    __syncthreads();
    float d = x - mean;
    red[t] = d * d;
    __syncthreads();
    for (int off = 256; off; off >>= 1) { if (t < off) red[t] += red[t + off]; __syncthreads(); }
    float rstd = rsqrtf(red[0] * (1.0f / MHID) + 1e-5f);
    out[t] = fmaxf(d * rstd * g[t] + bta[t], 0.0f);
}

// res[n] = sum_k hm[k]*mwo[k,n] + mbo[n] + tau[n]
__global__ void gemv2_kernel(const float* __restrict__ hm, const float* __restrict__ mwo,
                             const float* __restrict__ mbo, const float* __restrict__ tau,
                             float* __restrict__ out, int n) {
    __shared__ float sh[MHID];
    for (int k = threadIdx.x; k < MHID; k += blockDim.x) sh[k] = hm[k];
    __syncthreads();
    int nidx = blockIdx.x * blockDim.x + threadIdx.x;
    if (nidx >= n) return;
    float acc = mbo[nidx];
    #pragma unroll 8
    for (int k = 0; k < MHID; ++k) acc += sh[k] * mwo[(size_t)k * n + nidx];
    out[nidx] = acc + tau[nidx];
}

// ---------------- launch ----------------

extern "C" void kernel_launch(void* const* d_in, const int* in_sizes, int n_in,
                              void* d_out, int out_size, void* d_ws, size_t ws_size,
                              hipStream_t stream) {
    const float* tau   = (const float*)d_in[0];
    const int*   eidx  = (const int*)d_in[2];
    const float* ew    = (const float*)d_in[3];
    const float* gamma = (const float*)d_in[4];
    const float* lam   = (const float*)d_in[5];
    const float* fw1   = (const float*)d_in[7];
    const float* fb1   = (const float*)d_in[8];
    const float* fg1   = (const float*)d_in[9];
    const float* fbt1  = (const float*)d_in[10];
    const float* fw2   = (const float*)d_in[11];
    const float* fb2   = (const float*)d_in[12];
    const float* fg2   = (const float*)d_in[13];
    const float* fbt2  = (const float*)d_in[14];
    const float* fwo   = (const float*)d_in[15];
    const float* fbo   = (const float*)d_in[16];
    const float* mw1   = (const float*)d_in[17];
    const float* mb1   = (const float*)d_in[18];
    const float* mg1   = (const float*)d_in[19];
    const float* mbt1  = (const float*)d_in[20];
    const float* mwo   = (const float*)d_in[21];
    const float* mbo   = (const float*)d_in[22];
    float* out = (float*)d_out;

    const int N = in_sizes[0];       // 50000
    const int E = in_sizes[3];       // 1600000
    const int* row = eidx;
    const int* col = eidx + E;

    // workspace carve (floats)
    float* ws    = (float*)d_ws;
    float* deg   = ws;                 // N
    float* term  = ws + (size_t)N;     // N
    float* acc   = ws + (size_t)2 * N; // N  (== P)
    float* sacc  = ws + (size_t)3 * N; // N  (scatter temp)
    float* hat   = ws + (size_t)4 * N; // N
    float* u_nu  = ws + (size_t)5 * N; // 3N
    float* divv  = ws + (size_t)8 * N; // 3N
    float* hbufA = ws + (size_t)11 * N; // 64N
    float* hbufB = ws + (size_t)75 * N; // 64N
    float* prehm = ws + (size_t)139 * N; // 512
    float* hm    = prehm + MHID;         // 512

    const int TB = 256;
    int nBlkN  = (N + TB - 1) / TB;
    int nBlkE  = (E + TB - 1) / TB;
    int nBlk64 = (N * FH + TB - 1) / TB;

    // ---- heat-kernel Taylor expansion: P = exp(-BETA_DT * L) tau ----
    zero_kernel<<<nBlkN, TB, 0, stream>>>(deg, N);
    deg_kernel<<<nBlkE, TB, 0, stream>>>(row, ew, deg, E);
    init_term_acc<<<nBlkN, TB, 0, stream>>>(tau, term, acc, N);
    for (int kk = 1; kk <= K_TAYLOR_C; ++kk) {
        float coef = -BETA_DT_C / (float)kk;
        zero_kernel<<<nBlkN, TB, 0, stream>>>(sacc, N);
        spmv_edge<<<nBlkE, TB, 0, stream>>>(row, col, ew, term, sacc, E);
        taylor_node<<<nBlkN, TB, 0, stream>>>(deg, sacc, term, acc, coef, N);
    }

    // ---- per-node feature MLP (WMMA on the 64x64 hidden layer) ----
    layer1_kernel<<<nBlk64, TB, 0, stream>>>(tau, acc, gamma, lam, fw1, fb1, hbufA, N);
    ln_relu_64<<<(N + 7) / 8, 256, 0, stream>>>(hbufA, fg1, fbt1, hbufB, N);
    gemm_fw2_wmma<<<(N + 63) / 64, 128, 0, stream>>>(hbufB, fw2, fb2, hbufA, N);
    ln_relu_64<<<(N + 7) / 8, 256, 0, stream>>>(hbufA, fg2, fbt2, hbufB, N);
    layer_out_kernel<<<nBlkN, TB, 0, stream>>>(hbufB, fwo, fbo, tau, u_nu, N);

    // ---- flux divergence ----
    zero_kernel<<<(3 * N + TB - 1) / TB, TB, 0, stream>>>(divv, 3 * N);
    flux_div_kernel<<<nBlkE, TB, 0, stream>>>(row, col, ew, u_nu, divv, E);
    hat_kernel<<<nBlkN, TB, 0, stream>>>(tau, acc, divv, hat, N);

    // ---- mega-MLP: concat(3N) @ mw1 (Q block zero -> only 2N rows) ----
    zero_kernel<<<2, 256, 0, stream>>>(prehm, MHID);
    gemv1_kernel<<<(2 * N + 255) / 256, MHID, 0, stream>>>(hat, acc, mw1, prehm, N);
    ln_relu_512<<<1, MHID, 0, stream>>>(prehm, mb1, mg1, mbt1, hm);
    gemv2_kernel<<<nBlkN, TB, 0, stream>>>(hm, mwo, mbo, tau, out, N);

    (void)n_in; (void)out_size; (void)ws_size;
}
